// CenterEstimateNN_21191368639248
// MI455X (gfx1250) — matrix operand order repaired and checked by reading, not compiled, not run on hardware
//
#include <hip/hip_runtime.h>
#include <hip/hip_bf16.h>

// ==========================================================================
// PointCNN-style pipeline for MI455X (gfx1250, wave32).
// Heavy GEMMs -> v_wmma_f32_16x16x32_f16 (f16 in, f32 accum, fused epilogue),
// 128-bit fragment loads (A row-major contiguous, B pre-transposed to NxKp),
// one wave computes a 16x(NT*16) strip; NT is a template param so the
// unrolled WMMA loops carry no conditionals (no accumulator phi copies).
// Irregular per-point einsums / KNN / FPS -> VALU.
// ==========================================================================

typedef __attribute__((ext_vector_type(16))) _Float16 v16h;
typedef __attribute__((ext_vector_type(8)))  _Float16 v8h;
typedef __attribute__((ext_vector_type(8)))  float    v8f;

#define BATCH 16
#define NPTS  2048
#define N2    768
#define N3    256

static __device__ __forceinline__ float eluf(float x){ return x > 0.f ? x : (__expf(x) - 1.f); }

// ---------------------------------------------------------------------------
// f32 -> f16, zero-padded, row-major (for A / activations).
// ---------------------------------------------------------------------------
__global__ void cvt_pad_f16_kernel(const float* __restrict__ src, _Float16* __restrict__ dst,
                                   int R, int C, int Rp, int Cp){
  long id = (long)blockIdx.x*blockDim.x + threadIdx.x;
  long tot = (long)Rp*Cp;
  if(id >= tot) return;
  int r = (int)(id / Cp), c = (int)(id - (long)r*Cp);
  dst[id] = (r < R && c < C) ? (_Float16)src[(long)r*C + c] : (_Float16)0.0f;
}

// f32 (K x N) -> f16 transposed (N x Kp), K zero-padded to Kp (for B / weights).
__global__ void cvt_padT_f16_kernel(const float* __restrict__ src, _Float16* __restrict__ dst,
                                    int K, int N, int Kp){
  long id = (long)blockIdx.x*blockDim.x + threadIdx.x;
  long tot = (long)N*Kp;
  if(id >= tot) return;
  int n = (int)(id / Kp), k = (int)(id - (long)n*Kp);
  dst[id] = (k < K) ? (_Float16)src[(long)k*N + n] : (_Float16)0.0f;
}

// ---------------------------------------------------------------------------
// WMMA GEMM: C(MxN,f32) = A(MxKp,f16,row-major) * B (given as Bt: N x Kp, f16)
// Kp multiple of 32; M multiple of 64.
// Block = 128 thr (4 waves stacked on M); each wave computes 16 x (NT*16).
// One A fragment feeds NT back-to-back WMMAs per 32-wide K step.
// EPI: 0 none, 1 bias+ELU+bn(scale,shift), 2 bias+ReLU
// ---------------------------------------------------------------------------
template<int NT, int EPI>
__global__ __launch_bounds__(128) void wmma_gemm_kernel(
    const _Float16* __restrict__ A, const _Float16* __restrict__ Bt,
    float* __restrict__ C, int Kp, int N, int n0_base,
    const float* __restrict__ bias, const float* __restrict__ bns,
    const float* __restrict__ bnt){
  const int lane = threadIdx.x & 31;
  const int wave = threadIdx.x >> 5;
  const int tm  = blockIdx.y*64 + wave*16;
  const int tn0 = n0_base + blockIdx.x*(NT*16);
  const int l15 = lane & 15;            // A: M row / B,C: N col (per ISA 7.12.2)
  const int kg  = lane >> 4;            // K-half select

  const _Float16* Ap  = A  + (long)(tm + l15)*Kp + kg*8;
  const _Float16* Bp0 = Bt + (long)(tn0 + l15)*Kp + kg*8;

  v8f acc[NT];
  #pragma unroll
  for(int nt = 0; nt < NT; ++nt) acc[nt] = v8f{};

  for(int k0 = 0; k0 < Kp; k0 += 32){
    // A 16x32 fragment: two contiguous 16B runs per lane -> 2x b128
    v8h alo = *(const v8h*)(Ap + k0);
    v8h ahi = *(const v8h*)(Ap + k0 + 16);
    v16h a = __builtin_shufflevector(alo, ahi, 0,1,2,3,4,5,6,7,8,9,10,11,12,13,14,15);
    #pragma unroll
    for(int nt = 0; nt < NT; ++nt){     // no conditionals: acc stays pinned
      const _Float16* Bp = Bp0 + (long)nt*16*Kp + k0;
      v8h blo = *(const v8h*)(Bp);
      v8h bhi = *(const v8h*)(Bp + 16);
      v16h b = __builtin_shufflevector(blo, bhi, 0,1,2,3,4,5,6,7,8,9,10,11,12,13,14,15);
      acc[nt] = __builtin_amdgcn_wmma_f32_16x16x32_f16(
          false, a, false, b, (short)0, acc[nt], false, false);
    }
  }

  #pragma unroll
  for(int nt = 0; nt < NT; ++nt){
    const int col = tn0 + nt*16 + l15;
    float bi = 0.f, sc = 1.f, sh = 0.f;
    if(EPI != 0) bi = bias[col];
    if(EPI == 1){ sc = bns[col]; sh = bnt[col]; }
    #pragma unroll
    for(int r = 0; r < 8; ++r){         // C/D: VGPR r -> M = r + 8*kg, N = lane&15
      int row = tm + r + kg*8;
      float v = acc[nt][r];
      if(EPI == 1){ v += bi; v = eluf(v); v = v*sc + sh; }
      else if(EPI == 2){ v += bi; v = v > 0.f ? v : 0.f; }
      C[(long)row*N + col] = v;
    }
  }
}

// ---------------------------------------------------------------------------
// KNN: per query point keep K*DIL nearest (ascending d2, stable), emit every DILth.
// ---------------------------------------------------------------------------
template<int K, int DIL>
__global__ void knn_kernel(const float* __restrict__ pos, int n, int* __restrict__ nbr){
  int gid = blockIdx.x*blockDim.x + threadIdx.x;
  if(gid >= BATCH*n) return;
  int b = gid / n, i = gid - b*n;
  const float* pb = pos + (long)b*n*3;
  float px = pb[i*3+0], py = pb[i*3+1], pz = pb[i*3+2];
  constexpr int KD = K*DIL;
  float dist[KD]; int ind[KD];
  #pragma unroll
  for(int q = 0; q < KD; ++q){ dist[q] = 3.4e38f; ind[q] = 0; }
  for(int j = 0; j < n; ++j){
    float dx = pb[j*3+0]-px, dy = pb[j*3+1]-py, dz = pb[j*3+2]-pz;
    float d = dx*dx + dy*dy + dz*dz;
    if(d < dist[KD-1]){
      int p = KD-1;
      while(p > 0 && dist[p-1] > d){ dist[p] = dist[p-1]; ind[p] = ind[p-1]; --p; }
      dist[p] = d; ind[p] = j;
    }
  }
  int* out = nbr + (long)gid*K;
  #pragma unroll
  for(int q = 0; q < K; ++q) out[q] = ind[q*DIL];
}

// rel[b,i,j,:] = pos[b,nbr[b,i,j],:] - pos[b,i,:]
__global__ void rel_kernel(const float* __restrict__ pos, const int* __restrict__ nbr,
                           float* __restrict__ rel, int n, int k, long total){
  long id = (long)blockIdx.x*blockDim.x + threadIdx.x;
  if(id >= total) return;                   // total = B*n*k
  long b = id / ((long)n*k);
  long rem = id - b*(long)n*k;
  int i = (int)(rem / k);
  const float* pb = pos + b*(long)n*3;
  int nb = nbr[id];
  rel[id*3+0] = pb[nb*3+0] - pb[i*3+0];
  rel[id*3+1] = pb[nb*3+1] - pb[i*3+1];
  rel[id*3+2] = pb[nb*3+2] - pb[i*3+2];
}

// h0 = elu(rel @ w1a + b1a) * s + t     (K=3 -> cheap VALU)
__global__ void h0_kernel(const float* __restrict__ rel, const float* __restrict__ w,
                          const float* __restrict__ bias, const float* __restrict__ s,
                          const float* __restrict__ t, float* __restrict__ h0,
                          long M, int cd){
  long id = (long)blockIdx.x*blockDim.x + threadIdx.x;
  if(id >= M*cd) return;
  long m = id / cd; int c = (int)(id - m*cd);
  const float* r = rel + m*3;
  float v = bias[c] + r[0]*w[c] + r[1]*w[cd + c] + r[2]*w[2*cd + c];
  v = eluf(v);
  h0[id] = v*s[c] + t[c];
}

// h = concat([h1 (cd), gather(x_prev)[nbr] (c_in)], -1)
__global__ void concat_kernel(const float* __restrict__ h1, const float* __restrict__ x,
                              const int* __restrict__ nbr, float* __restrict__ h,
                              long Mk, int n, int k, int cd, int cin){
  int c = cd + cin;
  long id = (long)blockIdx.x*blockDim.x + threadIdx.x;
  if(id >= Mk*c) return;
  long m = id / c; int cc = (int)(id - m*c);
  if(cc < cd){ h[id] = h1[m*cd + cc]; }
  else{
    long b = m / ((long)n*k);
    int nb = nbr[m];
    h[id] = x[(b*(long)n + nb)*cin + (cc - cd)];
  }
}

// out[p,g,o] = act(sum_t in[p,g,t]*wc[g,o,t] + bc[g,o]) * s[g*k+o] + tsh[g*k+o]
__global__ void teinsum_kernel(const float* __restrict__ tin, const float* __restrict__ wc,
                               const float* __restrict__ bc, const float* __restrict__ s,
                               const float* __restrict__ tsh, float* __restrict__ tout,
                               long P, int k, int do_elu){
  long id = (long)blockIdx.x*blockDim.x + threadIdx.x;
  int kk = k*k;
  if(id >= P*kk) return;
  long p = id / kk; int rem = (int)(id - p*kk);
  int g = rem / k, o = rem - g*k;
  const float* tr = tin + p*(long)kk + g*k;
  const float* wr = wc + ((long)g*k + o)*k;
  float acc = bc[rem];
  for(int tt = 0; tt < k; ++tt) acc += tr[tt]*wr[tt];
  if(do_elu) acc = eluf(acc);
  tout[id] = acc*s[rem] + tsh[rem];
}

// y2[p,ci,m] = sum_l (sum_kk h[p,kk,ci]*t2[p,kk,l]) * wd[ci,m,l] + bd[ci,m]
__global__ void y_kernel(const float* __restrict__ h, const float* __restrict__ t2,
                         const float* __restrict__ wd, const float* __restrict__ bd,
                         float* __restrict__ y2, long P, int k, int c, int dm){
  long id = (long)blockIdx.x*blockDim.x + threadIdx.x;
  int cdm = c*dm;
  if(id >= P*cdm) return;
  long p = id / cdm; int rem = (int)(id - p*cdm);
  int ci = rem / dm, m = rem - ci*dm;
  const float* hp = h + p*(long)k*c;
  const float* tp = t2 + p*(long)k*k;
  const float* wdr = wd + ((long)ci*dm + m)*k;
  float acc = 0.f;
  for(int l = 0; l < k; ++l){
    float yl = 0.f;
    for(int q = 0; q < k; ++q) yl += hp[q*c + ci]*tp[q*k + l];
    acc += yl*wdr[l];
  }
  y2[id] = acc + bd[rem];
}

// ---------------------------------------------------------------------------
// Farthest point sampling: one block per batch, sequential scan w/ LDS reduce.
// Tie-break matches jnp.argmax (first occurrence).
// ---------------------------------------------------------------------------
__global__ __launch_bounds__(256) void fps_kernel(const float* __restrict__ pos,
                                                  int n, int m, int* __restrict__ idx){
  __shared__ float mind[NPTS];
  __shared__ float rv[256];
  __shared__ int   ri[256];
  __shared__ int   cur;
  int b = blockIdx.x, t = threadIdx.x;
  const float* pb = pos + (long)b*n*3;
  float x0 = pb[0], y0 = pb[1], z0 = pb[2];
  for(int j = t; j < n; j += 256){
    float dx = pb[j*3]-x0, dy = pb[j*3+1]-y0, dz = pb[j*3+2]-z0;
    mind[j] = dx*dx + dy*dy + dz*dz;
  }
  if(t == 0) idx[(long)b*m] = 0;
  __syncthreads();
  for(int s = 1; s < m; ++s){
    float bv = -1.f; int bi = n;
    for(int j = t; j < n; j += 256){
      float v = mind[j];
      if(v > bv || (v == bv && j < bi)){ bv = v; bi = j; }
    }
    rv[t] = bv; ri[t] = bi;
    __syncthreads();
    for(int off = 128; off > 0; off >>= 1){
      if(t < off){
        float v2 = rv[t+off]; int i2 = ri[t+off];
        if(v2 > rv[t] || (v2 == rv[t] && i2 < ri[t])){ rv[t] = v2; ri[t] = i2; }
      }
      __syncthreads();
    }
    if(t == 0){ cur = ri[0]; idx[(long)b*m + s] = cur; }
    __syncthreads();
    int ci = cur;
    float cx = pb[ci*3], cy = pb[ci*3+1], cz = pb[ci*3+2];
    for(int j = t; j < n; j += 256){
      float dx = pb[j*3]-cx, dy = pb[j*3+1]-cy, dz = pb[j*3+2]-cz;
      float d = dx*dx + dy*dy + dz*dz;
      if(d < mind[j]) mind[j] = d;
    }
    __syncthreads();
  }
}

__global__ void gather_rows_kernel(const float* __restrict__ src, const int* __restrict__ idx,
                                   float* __restrict__ dst, int n_old, int m, int c){
  long id = (long)blockIdx.x*blockDim.x + threadIdx.x;
  long tot = (long)BATCH*m*c;
  if(id >= tot) return;
  long b = id / ((long)m*c);
  long rem = id - b*(long)m*c;
  int i = (int)(rem / c), cc = (int)(rem - (long)i*c);
  dst[id] = src[(b*(long)n_old + idx[b*m + i])*c + cc];
}

// Head: mean over N3 points -> concat onehot -> 3 dense layers. One block/batch.
__global__ __launch_bounds__(256) void head_kernel(
    const float* __restrict__ x3, const float* __restrict__ cls,
    const float* __restrict__ wl1, const float* __restrict__ bl1,
    const float* __restrict__ wl2, const float* __restrict__ bl2,
    const float* __restrict__ wl3, const float* __restrict__ bl3,
    float* __restrict__ out){
  __shared__ float h0s[200], h1s[256], h2s[128];
  int b = blockIdx.x, t = threadIdx.x;
  if(t < 192){
    float s = 0.f;
    for(int i = 0; i < N3; ++i) s += x3[((long)b*N3 + i)*192 + t];
    h0s[t] = s / (float)N3;
  } else if(t < 200){
    h0s[t] = cls[b*8 + (t - 192)];
  }
  __syncthreads();
  { float a = bl1[t]; for(int j = 0; j < 200; ++j) a += h0s[j]*wl1[j*256 + t];
    h1s[t] = a > 0.f ? a : 0.f; }
  __syncthreads();
  if(t < 128){ float a = bl2[t]; for(int j = 0; j < 256; ++j) a += h1s[j]*wl2[j*128 + t];
    h2s[t] = a > 0.f ? a : 0.f; }
  __syncthreads();
  if(t < 3){ float a = bl3[t]; for(int j = 0; j < 128; ++j) a += h2s[j]*wl3[j*3 + t];
    out[b*3 + t] = a; }
}

// ==========================================================================
// Host-side orchestration
// ==========================================================================
struct XParams {
  const float *w1a,*b1a,*bn1a_s,*bn1a_t,*w1b,*b1b,*bn1b_s,*bn1b_t;
  const float *w2a,*b2a,*bn2a_s,*bn2a_t,*wc1,*bc1,*bn2b_s,*bn2b_t;
  const float *wc2,*bc2,*bn2c_s,*bn2c_t,*wd,*bd,*wf,*bf;
};

struct Scratch {
  int* nbr; float* rel; _Float16* prh;
  float* h0; _Float16* h0h; float* h1; float* hcat;
  float* t0; float* t1; float* t2;
  float* y2; _Float16* y2h; _Float16* wB;
};

static inline int grid1d(long total){ return (int)((total + 255) / 256); }

static void launch_cvt(const float* src, _Float16* dst, int R, int C, int Rp, int Cp,
                       hipStream_t st){
  long tot = (long)Rp*Cp;
  cvt_pad_f16_kernel<<<grid1d(tot), 256, 0, st>>>(src, dst, R, C, Rp, Cp);
}

static void launch_cvtT(const float* src, _Float16* dst, int K, int N, int Kp,
                        hipStream_t st){
  long tot = (long)N*Kp;
  cvt_padT_f16_kernel<<<grid1d(tot), 256, 0, st>>>(src, dst, K, N, Kp);
}

template<int NT>
static void launch_gemm_nt(const _Float16* A, const _Float16* Bt, float* C,
                           long M, int Kp, int N, int gx, int base,
                           const float* bias, const float* s, const float* t,
                           int epi, hipStream_t st){
  dim3 grid((unsigned)gx, (unsigned)(M/64));
  if(epi == 1)
    wmma_gemm_kernel<NT,1><<<grid, 128, 0, st>>>(A, Bt, C, Kp, N, base, bias, s, t);
  else if(epi == 2)
    wmma_gemm_kernel<NT,2><<<grid, 128, 0, st>>>(A, Bt, C, Kp, N, base, bias, s, t);
  else
    wmma_gemm_kernel<NT,0><<<grid, 128, 0, st>>>(A, Bt, C, Kp, N, base, bias, s, t);
}

static void launch_gemm(const _Float16* A, const _Float16* Bt, float* C,
                        long M, int Kp, int N, const float* bias,
                        const float* s, const float* t, int epi, hipStream_t st){
  int full = N/64;                // number of 64-wide strips (NT=4)
  int rem  = (N - full*64)/16;    // leftover 16-wide tiles (NT=1..3)
  if(full > 0) launch_gemm_nt<4>(A, Bt, C, M, Kp, N, full, 0, bias, s, t, epi, st);
  if(rem == 1) launch_gemm_nt<1>(A, Bt, C, M, Kp, N, 1, full*64, bias, s, t, epi, st);
  else if(rem == 2) launch_gemm_nt<2>(A, Bt, C, M, Kp, N, 1, full*64, bias, s, t, epi, st);
  else if(rem == 3) launch_gemm_nt<3>(A, Bt, C, M, Kp, N, 1, full*64, bias, s, t, epi, st);
}

static void run_xconv(const XParams& P, const float* x_in, int c_in,
                      const float* pos, int n, int k, int dil,
                      int cd, int c_out, int dm,
                      float* x_out, Scratch& W, hipStream_t st){
  const int c = c_in + cd;
  const long Pn = (long)BATCH*n;     // points     (multiple of 64 for all layers)
  const long Mk = Pn*k;              // point-neighbor pairs (multiple of 64)

  // KNN (dilated)
  {
    int total = BATCH*n, blk = 128, grd = (total + blk - 1)/blk;
    if(k == 8)       knn_kernel<8, 1><<<grd, blk, 0, st>>>(pos, n, W.nbr);
    else if(k == 12) knn_kernel<12,2><<<grd, blk, 0, st>>>(pos, n, W.nbr);
    else             knn_kernel<16,2><<<grd, blk, 0, st>>>(pos, n, W.nbr);
  }
  rel_kernel<<<grid1d(Mk), 256, 0, st>>>(pos, W.nbr, W.rel, n, k, Mk);

  // h branch: 3->cd (VALU), cd->cd (WMMA, fused ELU+BN)
  h0_kernel<<<grid1d(Mk*cd), 256, 0, st>>>(W.rel, P.w1a, P.b1a, P.bn1a_s, P.bn1a_t,
                                           W.h0, Mk, cd);
  launch_cvt(W.h0, W.h0h, (int)Mk, cd, (int)Mk, cd, st);
  launch_cvtT(P.w1b, W.wB, cd, cd, cd, st);
  launch_gemm(W.h0h, W.wB, W.h1, Mk, cd, cd, P.b1b, P.bn1b_s, P.bn1b_t, /*epi=*/1, st);

  const float* h = W.h1;
  if(c_in > 0){
    concat_kernel<<<grid1d(Mk*c), 256, 0, st>>>(W.h1, x_in, W.nbr, W.hcat,
                                                Mk, n, k, cd, c_in);
    h = W.hcat;
  }

  // transform branch: pr@w2a (WMMA), then two kxk learned-transform einsums (VALU)
  const int K3 = 3*k, Kp3 = ((K3 + 31)/32)*32, kk = k*k;
  launch_cvt(W.rel, W.prh, (int)Pn, K3, (int)Pn, Kp3, st);   // rel viewed as (Pn, 3k)
  launch_cvtT(P.w2a, W.wB, K3, kk, Kp3, st);
  launch_gemm(W.prh, W.wB, W.t0, Pn, Kp3, kk, P.b2a, P.bn2a_s, P.bn2a_t, /*epi=*/1, st);
  teinsum_kernel<<<grid1d(Pn*kk), 256, 0, st>>>(W.t0, P.wc1, P.bc1, P.bn2b_s, P.bn2b_t,
                                                W.t1, Pn, k, /*elu=*/1);
  teinsum_kernel<<<grid1d(Pn*kk), 256, 0, st>>>(W.t1, P.wc2, P.bc2, P.bn2c_s, P.bn2c_t,
                                                W.t2, Pn, k, /*elu=*/0);

  // y = t^T h contracted with wd (VALU), then wf GEMM (WMMA, fused bias+ReLU)
  const int cdm = c*dm, Kpf = ((cdm + 31)/32)*32;
  y_kernel<<<grid1d(Pn*cdm), 256, 0, st>>>(h, W.t2, P.wd, P.bd, W.y2, Pn, k, c, dm);
  launch_cvt(W.y2, W.y2h, (int)Pn, cdm, (int)Pn, Kpf, st);
  launch_cvtT(P.wf, W.wB, cdm, c_out, Kpf, st);
  launch_gemm(W.y2h, W.wB, x_out, Pn, Kpf, c_out, P.bf, nullptr, nullptr, /*epi=*/2, st);
}

static void fill_xparams(XParams& X, void* const* d_in, int base){
  const float** p = (const float**)&X;
  for(int i = 0; i < 24; ++i) p[i] = (const float*)d_in[base + i];
}

extern "C" void kernel_launch(void* const* d_in, const int* in_sizes, int n_in,
                              void* d_out, int out_size, void* d_ws, size_t ws_size,
                              hipStream_t stream) {
  (void)in_sizes; (void)n_in; (void)out_size; (void)ws_size;
  // Input order = setup_inputs() dict insertion order, params pytree flattened
  // in construction order: x1(24), x2(24), x3(24), wl1, bl1, wl2, bl2, wl3, bl3,
  // then pos, batch, cls_onehot.
  XParams X1, X2, X3;
  fill_xparams(X1, d_in, 0);
  fill_xparams(X2, d_in, 24);
  fill_xparams(X3, d_in, 48);
  const float* wl1 = (const float*)d_in[72];
  const float* bl1 = (const float*)d_in[73];
  const float* wl2 = (const float*)d_in[74];
  const float* bl2 = (const float*)d_in[75];
  const float* wl3 = (const float*)d_in[76];
  const float* bl3 = (const float*)d_in[77];
  const float* pos_in = (const float*)d_in[78];   // (B*NPTS, 3) == (B, NPTS, 3)
  const float* cls    = (const float*)d_in[80];   // (B, 8); d_in[79]=batch unused

  // ---- carve workspace (bump allocator, 256B aligned) ----
  char* w = (char*)d_ws;
  auto alloc = [&](size_t bytes) -> void* {
    void* p = (void*)w; w += (bytes + 255) & ~(size_t)255; return p;
  };
  float* posA = (float*)alloc((size_t)BATCH*NPTS*3*4);
  float* posB = (float*)alloc((size_t)BATCH*NPTS*3*4);
  float* xA   = (float*)alloc((size_t)BATCH*NPTS*48*4);   // big enough for all layers
  float* xB   = (float*)alloc((size_t)BATCH*NPTS*48*4);
  int*   idxb = (int*)  alloc((size_t)BATCH*N2*4);

  Scratch W;
  W.nbr  = (int*)      alloc((size_t)BATCH*NPTS*8*4);          // max b*n*k
  W.rel  = (float*)    alloc((size_t)BATCH*NPTS*8*3*4);        // max b*n*k*3
  W.prh  = (_Float16*) alloc((size_t)BATCH*NPTS*64*2);         // max Pn*Kp3
  W.h0   = (float*)    alloc((size_t)9437184*4);               // max Mk*cd (layer2)
  W.h0h  = (_Float16*) alloc((size_t)9437184*2);
  W.h1   = (float*)    alloc((size_t)9437184*4);
  W.hcat = (float*)    alloc((size_t)16515072*4);              // max Mk*c (layer2)
  W.t0   = (float*)    alloc((size_t)2097152*4);               // max Pn*k*k (layer1)
  W.t1   = (float*)    alloc((size_t)2097152*4);
  W.t2   = (float*)    alloc((size_t)2097152*4);
  W.y2   = (float*)    alloc((size_t)2097152*4);               // max Pn*c*dm (layer1)
  W.y2h  = (_Float16*) alloc((size_t)2097152*2);
  W.wB   = (_Float16*) alloc((size_t)65536*2);                 // max padded weight (NxKp)

  // ---- layer 1: n=2048, k=8, dil=1, cd=32, c_in=0 -> c_out=48 (dm=2) ----
  run_xconv(X1, nullptr, 0, pos_in, NPTS, 8, 1, 32, 48, 2, xA, W, stream);

  // FPS 2048 -> 768, gather
  fps_kernel<<<BATCH, 256, 0, stream>>>(pos_in, NPTS, N2, idxb);
  gather_rows_kernel<<<grid1d((long)BATCH*N2*48), 256, 0, stream>>>(xA, idxb, xB, NPTS, N2, 48);
  gather_rows_kernel<<<grid1d((long)BATCH*N2*3), 256, 0, stream>>>(pos_in, idxb, posA, NPTS, N2, 3);

  // ---- layer 2: n=768, k=12, dil=2, cd=64, c_in=48 -> c_out=96 (dm=1) ----
  run_xconv(X2, xB, 48, posA, N2, 12, 2, 64, 96, 1, xA, W, stream);

  // FPS 768 -> 256, gather
  fps_kernel<<<BATCH, 256, 0, stream>>>(posA, N2, N3, idxb);
  gather_rows_kernel<<<grid1d((long)BATCH*N3*96), 256, 0, stream>>>(xA, idxb, xB, N2, N3, 96);
  gather_rows_kernel<<<grid1d((long)BATCH*N3*3), 256, 0, stream>>>(posA, idxb, posB, N2, N3, 3);

  // ---- layer 3: n=256, k=16, dil=2, cd=128, c_in=96 -> c_out=192 (dm=1) ----
  run_xconv(X3, xB, 96, posB, N3, 16, 2, 128, 192, 1, xA, W, stream);

  // ---- head: mean + MLP -> (16, 3) ----
  head_kernel<<<BATCH, 256, 0, stream>>>(xA, cls, wl1, bl1, wl2, bl2, wl3, bl3,
                                         (float*)d_out);
}
